// DFT_66314295050841
// MI455X (gfx1250) — matrix operand order repaired and checked by reading, not compiled, not run on hardware
//
#include <hip/hip_runtime.h>
#include <hip/hip_bf16.h>
#include <math.h>

// B=8, C=256, O=256, H=64, W=64, K=3, K2=9, Kdim = C*K2 = 2304
// Flattened GEMM: out[M=32768, N=256] = A[M, 2304] * B[2304, 256]
//   A[m][k*256+c] = bilinear(x; m, tap k) * mask[m,k]
//   B[k*256+c][o] = w_proj[o][c][k]

typedef __attribute__((ext_vector_type(8)))  float   v8f;
typedef __attribute__((ext_vector_type(8)))  __bf16  v8bf;
typedef __attribute__((ext_vector_type(16))) __bf16  v16bf;
typedef __attribute__((ext_vector_type(4)))  unsigned int u32x4;
typedef __attribute__((ext_vector_type(8)))  int     i32x8;
typedef __attribute__((ext_vector_type(4)))  int     i32x4;

#define KSTEPS 72          // 2304 / 32
#define LDA    40          // padded LDS row stride in halves (80B -> conflict free)

#ifndef __has_builtin
#define __has_builtin(x) 0
#endif
#if __has_builtin(__builtin_amdgcn_tensor_load_to_lds)
#define USE_TDM 1
#else
#define USE_TDM 0
#endif

// ---------------------------------------------------------------------------
// Kernel 1: reorder w_proj (fp32 [O][C][3][3]) into bf16 tiles [s][o][j],
// s = K-step (32 K-values), flat K index kc = k*256 + c, j = kc % 32.
// Each step tile is 16KB contiguous -> ideal TDM 1-D descriptor.
// ---------------------------------------------------------------------------
__global__ __launch_bounds__(256) void prep_weights_kernel(
    const float* __restrict__ w_proj, unsigned short* __restrict__ wkraw)
{
    __bf16* wk = reinterpret_cast<__bf16*>(wkraw);
    int idx = blockIdx.x * 256 + threadIdx.x;       // < 72*256*32 = 589824
    int s   = idx >> 13;                            // / 8192
    int rem = idx & 8191;
    int o   = rem >> 5;
    int j   = rem & 31;
    int kc  = (s << 5) + j;
    int k   = kc >> 8;                              // tap index 0..8
    int c   = kc & 255;
    wk[idx] = (__bf16)w_proj[((size_t)(o * 256 + c)) * 9 + k];
}

// ---------------------------------------------------------------------------
// Kernel 2: offset + mask convs (27 output channels, 3x3 over C=256), fp32.
// ---------------------------------------------------------------------------
__global__ __launch_bounds__(256) void offmask_kernel(
    const float* __restrict__ x,      const float* __restrict__ w_off,
    const float* __restrict__ b_off,  const float* __restrict__ w_mask,
    const float* __restrict__ b_mask, float* __restrict__ dyb,
    float* __restrict__ dxb,          float* __restrict__ mkb)
{
    int m  = blockIdx.x * 256 + threadIdx.x;        // 0..32767
    int b  = m >> 12;
    int hw = m & 4095;
    int h  = hw >> 6;
    int w  = hw & 63;
    const float* xb = x + ((size_t)b << 20);        // b*256*4096

    float acc[27];
#pragma unroll
    for (int i = 0; i < 27; ++i) acc[i] = 0.f;

    for (int c = 0; c < 256; ++c) {
        const float* xc = xb + ((size_t)c << 12);
        float n[9];
#pragma unroll
        for (int j = 0; j < 9; ++j) {
            int hh = h + j / 3 - 1;
            int ww = w + j % 3 - 1;
            bool v = (hh >= 0) && (hh < 64) && (ww >= 0) && (ww < 64);
            n[j] = v ? xc[hh * 64 + ww] : 0.f;
        }
#pragma unroll
        for (int oc = 0; oc < 18; ++oc) {
            const float* wr = w_off + ((size_t)(oc * 256 + c)) * 9;
#pragma unroll
            for (int j = 0; j < 9; ++j) acc[oc] += n[j] * wr[j];
        }
#pragma unroll
        for (int oc = 0; oc < 9; ++oc) {
            const float* wr = w_mask + ((size_t)(oc * 256 + c)) * 9;
#pragma unroll
            for (int j = 0; j < 9; ++j) acc[18 + oc] += n[j] * wr[j];
        }
    }

#pragma unroll
    for (int k = 0; k < 9; ++k) {
        float dv  = acc[2 * k]     + b_off[2 * k];
        float dxv = acc[2 * k + 1] + b_off[2 * k + 1];
        dv  = fminf(fmaxf(dv,  -16.f), 16.f);       // max_off = min(H,W)//4
        dxv = fminf(fmaxf(dxv, -16.f), 16.f);
        float mv = 2.f / (1.f + expf(-(acc[18 + k] + b_mask[k])));
        size_t idx = ((size_t)(b * 9 + k) << 12) + hw;
        dyb[idx] = dv;
        dxb[idx] = dxv;
        mkb[idx] = mv;
    }
}

// ---------------------------------------------------------------------------
// TDM helper: DMA one 16KB weight tile (2048 x 8B elements) into LDS with
// 16B padding inserted after every 64B row -> padded stride LDA=40 halves.
// Descriptor per CDNA5 ISA (D# groups 0/1; rest zero).
// clang-23 toolchain: 6-arg builtin (g0 u32x4, g1 i32x8, i32x4, i32x4, i32x8, cpol)
// ---------------------------------------------------------------------------
#if USE_TDM
__device__ __forceinline__ void tdm_load_wtile(const __bf16* wk, int s,
                                               unsigned lds_addr)
{
    unsigned long long ga =
        (unsigned long long)(const void*)wk + ((unsigned long long)s << 14);

    u32x4 g0;
    g0[0] = 1u;                                      // count=1 (valid user D#)
    g0[1] = lds_addr;                                // LDS byte address
    g0[2] = (unsigned)(ga & 0xFFFFFFFFu);            // global_addr[31:0]
    g0[3] = (unsigned)((ga >> 32) & 0x01FFFFFFu)     // global_addr[56:32]
          | (2u << 30);                              // type = 2 ("image")

    i32x8 g1;
    g1[0] = (3 << 16)                                // data_size = 8 bytes
          | (1 << 20)                                // pad_enable
          | (3 << 22)                                // pad_interval: 16 DW (64B)
          | (3 << 25);                               // pad_amount:    4 DW (16B)
    g1[1] = (2048 << 16);                            // tensor_dim0[15:0]
    g1[2] = (1 << 16);                               // dim0 hi=0 | tensor_dim1=1
    g1[3] = (2048 << 16);                            // dim1 hi=0 | tile_dim0=2048
    g1[4] = 1;                                       // tile_dim1=1, tile_dim2=0
    g1[5] = 2048;                                    // tensor_dim0_stride[31:0]
    g1[6] = 0;                                       // stride hi | dim1_stride lo
    g1[7] = 0;

    i32x4 gz4 = {0, 0, 0, 0};
    i32x8 gz8 = {0, 0, 0, 0, 0, 0, 0, 0};
    __builtin_amdgcn_tensor_load_to_lds(g0, g1, gz4, gz4, gz8, 0);
}
#endif

// ---------------------------------------------------------------------------
// Kernel 3: fused bilinear-sample + projection GEMM via WMMA bf16.
// 256 threads = 8 waves; workgroup tile = 32 pixels x 256 outputs.
// Wave w: rows m0=(w>>2)*16, cols n0=(w&3)*64 -> 4 accumulators of 16x16.
// Weight tiles stream through double-buffered LDS via the TDM (TENSORcnt).
// ---------------------------------------------------------------------------
__global__ __launch_bounds__(256) void dft_gemm_kernel(
    const float* __restrict__ x,    const float* __restrict__ dyb,
    const float* __restrict__ dxb,  const float* __restrict__ mkb,
    const unsigned short* __restrict__ wkraw,
    const float* __restrict__ b_proj, float* __restrict__ outp)
{
    const __bf16* wk = reinterpret_cast<const __bf16*>(wkraw);

    __shared__ __align__(16) __bf16 sA[32 * LDA];       //  2.5 KB
    __shared__ __align__(16) __bf16 sB[2][256 * LDA];   // 40.0 KB (double buf)

    const int t    = threadIdx.x;
    const int lane = t & 31;
    const int wave = t >> 5;
    const int m0   = (wave >> 2) * 16;
    const int n0   = (wave & 3) * 64;
    const int tileBase = blockIdx.x * 32;

    // gather mapping: thread owns pixel r = t&31, 4 channel columns jb..jb+3
    const int r  = t & 31;
    const int jb = (t >> 5) * 4;
    const int m  = tileBase + r;
    const int b  = m >> 12;
    const int hw = m & 4095;
    const int h  = hw >> 6;
    const int wv = hw & 63;
    const float* xb = x + ((size_t)b << 20);

    v8f acc[4] = {};
    int buf = 0;

#if USE_TDM
    // flat shared pointer truncated to 32 bits == LDS byte offset (ISA 10.2)
    unsigned ldsb0 = (unsigned)(size_t)&sB[0][0];
    unsigned ldsb1 = (unsigned)(size_t)&sB[1][0];
    if (wave == 0) tdm_load_wtile(wk, 0, ldsb0);
#endif

    for (int k = 0; k < 9; ++k) {
        // --- per-pixel bilinear setup for tap k (weights premultiplied by mask)
        size_t oidx = ((size_t)(b * 9 + k) << 12) + hw;
        float dyv = dyb[oidx];
        float dxv = dxb[oidx];
        float mv  = mkb[oidx];
        float sy  = (float)(h + (k / 3) - 1) + dyv;
        float sx  = (float)(wv + (k % 3) - 1) + dxv;
        float fy  = floorf(sy), fx = floorf(sx);
        int   y0  = (int)fy,    x0 = (int)fx;
        int   y1  = y0 + 1,     x1 = x0 + 1;
        float ty  = sy - fy,    tx = sx - fx;
        float w00 = (1.f - ty) * (1.f - tx) * mv;
        float w01 = (1.f - ty) * tx * mv;
        float w10 = ty * (1.f - tx) * mv;
        float w11 = ty * tx * mv;
        bool vy0 = (y0 >= 0) & (y0 < 64), vy1 = (y1 >= 0) & (y1 < 64);
        bool vx0 = (x0 >= 0) & (x0 < 64), vx1 = (x1 >= 0) & (x1 < 64);
        w00 = (vy0 && vx0) ? w00 : 0.f;
        w01 = (vy0 && vx1) ? w01 : 0.f;
        w10 = (vy1 && vx0) ? w10 : 0.f;
        w11 = (vy1 && vx1) ? w11 : 0.f;
        int cy0 = min(max(y0, 0), 63), cy1 = min(max(y1, 0), 63);
        int cx0 = min(max(x0, 0), 63), cx1 = min(max(x1, 0), 63);
        int o00 = cy0 * 64 + cx0, o01 = cy0 * 64 + cx1;
        int o10 = cy1 * 64 + cx0, o11 = cy1 * 64 + cx1;

        for (int cc = 0; cc < 8; ++cc) {
            const int s = k * 8 + cc;
            __syncthreads();   // prior step's reads of sA / sB[buf^1] are done

            // --- fill A tile: 4 bilinear samples per thread
            const int c0 = (cc << 5) + jb;
#pragma unroll
            for (int j = 0; j < 4; ++j) {
                const int coff = (c0 + j) << 12;
                float v = w00 * xb[coff + o00] + w01 * xb[coff + o01]
                        + w10 * xb[coff + o10] + w11 * xb[coff + o11];
                sA[r * LDA + jb + j] = (__bf16)v;
            }

#if USE_TDM
            // --- wave0: kick DMA of next tile, guarantee current tile landed
            if (wave == 0) {
                if (s + 1 < KSTEPS) {
                    tdm_load_wtile(wk, s + 1, (buf ^ 1) ? ldsb1 : ldsb0);
                    __builtin_amdgcn_s_wait_tensorcnt(1);  // in-order: tile s done
                } else {
                    __builtin_amdgcn_s_wait_tensorcnt(0);
                }
            }
#else
            // --- fallback: synchronous cooperative B-tile load (row o = t)
            const uint4* g = (const uint4*)(wk + (((size_t)s << 8) + t) * 32);
            uint4 wlo = g[0];
            uint4 whi = g[1];
            *((uint4*)&sB[buf][t * LDA])     = wlo;
            *((uint4*)&sB[buf][t * LDA + 8]) = whi;
            if (s + 1 < KSTEPS)
                __builtin_prefetch(wk + (((size_t)(s + 1) << 8) + t) * 32, 0, 0);
#endif
            __syncthreads();   // publish sA and sB[buf]

            // --- A fragment: lane L holds M=L%16, K=(L/16)*16..+15
            v16bf afrag;
            {
                const int arow = (m0 + (lane & 15)) * LDA;
                const int kb   = (lane >> 4) * 16;
                v8bf alo = *(const v8bf*)&sA[arow + kb];
                v8bf ahi = *(const v8bf*)&sA[arow + kb + 8];
#pragma unroll
                for (int i = 0; i < 8; ++i) { afrag[i] = alo[i]; afrag[i + 8] = ahi[i]; }
            }

            // --- 4 N-subtiles: B fragment lane L holds N=L%16, K=(L/16)*16..+15
#pragma unroll
            for (int nt = 0; nt < 4; ++nt) {
                const int nrow = (n0 + nt * 16 + (lane & 15)) * LDA;
                const int kb2  = (lane >> 4) * 16;
                v8bf blo = *(const v8bf*)&sB[buf][nrow + kb2];
                v8bf bhi = *(const v8bf*)&sB[buf][nrow + kb2 + 8];
                v16bf bfrag;
#pragma unroll
                for (int i = 0; i < 8; ++i) { bfrag[i] = blo[i]; bfrag[i + 8] = bhi[i]; }

                acc[nt] = __builtin_amdgcn_wmma_f32_16x16x32_bf16(
                    false, afrag, false, bfrag, (short)0, acc[nt], false, false);
            }
            buf ^= 1;
        }
    }

    // --- epilogue: + b_proj, scatter to out_pre[b][o][hw]
#pragma unroll
    for (int nt = 0; nt < 4; ++nt) {
        const int o  = n0 + nt * 16 + (lane & 15);
        const float bp = b_proj[o];
#pragma unroll
        for (int g8 = 0; g8 < 8; ++g8) {
            const int mrow = m0 + g8 + ((lane >> 4) << 3);
            const int mm   = tileBase + mrow;
            const int bb   = mm >> 12;
            const int hw2  = mm & 4095;
            outp[((size_t)((bb << 8) + o) << 12) + hw2] = acc[nt][g8] + bp;
        }
    }
}

// ---------------------------------------------------------------------------
// Kernel 4: per-channel mean / inverse-std over (B,H,W) = 32768 elements.
// ---------------------------------------------------------------------------
__global__ __launch_bounds__(256) void stats_kernel(
    const float* __restrict__ outp, float* __restrict__ meanb,
    float* __restrict__ istdb)
{
    const int o = blockIdx.x;
    const int t = threadIdx.x;
    float s = 0.f, s2 = 0.f;
    for (int i = t; i < 32768; i += 256) {
        int b  = i >> 12;
        int hw = i & 4095;
        float v = outp[((size_t)((b << 8) + o) << 12) + hw];
        s  += v;
        s2 += v * v;
    }
    __shared__ float sh[256];
    __shared__ float sh2[256];
    sh[t] = s; sh2[t] = s2;
    __syncthreads();
    for (int st = 128; st > 0; st >>= 1) {
        if (t < st) { sh[t] += sh[t + st]; sh2[t] += sh2[t + st]; }
        __syncthreads();
    }
    if (t == 0) {
        float mu  = sh[0] * (1.f / 32768.f);
        float var = sh2[0] * (1.f / 32768.f) - mu * mu;
        meanb[o]  = mu;
        istdb[o]  = rsqrtf(var + 1e-5f);
    }
}

// ---------------------------------------------------------------------------
// Kernel 5: normalize + gamma/beta + exact GELU.
// ---------------------------------------------------------------------------
__global__ __launch_bounds__(256) void norm_gelu_kernel(
    const float* __restrict__ outp, const float* __restrict__ meanb,
    const float* __restrict__ istdb, const float* __restrict__ gamma,
    const float* __restrict__ beta, float* __restrict__ out)
{
    size_t idx = (size_t)blockIdx.x * 256 + threadIdx.x;   // < 8*256*4096
    int o = (int)((idx >> 12) & 255);
    float v = outp[idx];
    float y = (v - meanb[o]) * istdb[o] * gamma[o] + beta[o];
    out[idx] = 0.5f * y * (1.f + erff(y * 0.70710678118654752f));
}

// ---------------------------------------------------------------------------
extern "C" void kernel_launch(void* const* d_in, const int* in_sizes, int n_in,
                              void* d_out, int out_size, void* d_ws, size_t ws_size,
                              hipStream_t stream)
{
    const float* x      = (const float*)d_in[0];
    const float* w_off  = (const float*)d_in[1];
    const float* b_off  = (const float*)d_in[2];
    const float* w_mask = (const float*)d_in[3];
    const float* b_mask = (const float*)d_in[4];
    const float* w_proj = (const float*)d_in[5];
    const float* b_proj = (const float*)d_in[6];
    const float* gamma  = (const float*)d_in[7];
    const float* beta   = (const float*)d_in[8];
    float* out = (float*)d_out;

    char* ws = (char*)d_ws;
    size_t off = 0;
    unsigned short* wkraw = (unsigned short*)(ws + off); off += 1179648;
    float* dyb   = (float*)(ws + off); off += 1179648;
    float* dxb   = (float*)(ws + off); off += 1179648;
    float* mkb   = (float*)(ws + off); off += 1179648;
    float* outp  = (float*)(ws + off); off += 33554432;
    float* meanb = (float*)(ws + off); off += 1024;
    float* istdb = (float*)(ws + off); off += 1024;

    prep_weights_kernel<<<2304, 256, 0, stream>>>(w_proj, wkraw);
    offmask_kernel<<<128, 256, 0, stream>>>(x, w_off, b_off, w_mask, b_mask,
                                            dyb, dxb, mkb);
    dft_gemm_kernel<<<1024, 256, 0, stream>>>(x, dyb, dxb, mkb, wkraw, b_proj, outp);
    stats_kernel<<<256, 256, 0, stream>>>(outp, meanb, istdb);
    norm_gelu_kernel<<<32768, 256, 0, stream>>>(outp, meanb, istdb, gamma, beta, out);
}
